// Model_51616916964130
// MI455X (gfx1250) — matrix operand order repaired and checked by reading, not compiled, not run on hardware
//
#include <hip/hip_runtime.h>
#include <hip/hip_bf16.h>
#include <math.h>

// ---------------- problem constants (match reference) ----------------
#define BS    32
#define TLEN  512
#define GG    (BS * TLEN)     // 16384 graphs / sequence rows
#define NNODE 64
#define NEDGE 126
#define INF_  128             // encoder input size  (N*2)
#define H8    1024            // encoder hidden
#define HOUT  128             // decoder hidden / output size

// ---------------- WMMA types (gfx1250, wave32) ----------------
typedef __attribute__((ext_vector_type(16))) __bf16 v16bf;
typedef __attribute__((ext_vector_type(8)))  float  v8f;

union FragBF { v16bf v; uint4 q[2]; };

__device__ __forceinline__ unsigned short f2bf(float f) {
  unsigned int u = __float_as_uint(f);
  u += 0x7FFFu + ((u >> 16) & 1u);          // round-to-nearest-even
  return (unsigned short)(u >> 16);
}

// A fragment (16x32 bf16, MxK): lane L -> row M = L%16, half = L/16 covers
// K in [8h, 8h+8) and [16+8h, 16+8h+8)  (ISA 7.12.2, 16-bit A 16x32).
__device__ __forceinline__ v16bf load_a_frag(const unsigned short* A, size_t row,
                                             int ldk, int k0, int half) {
  FragBF f;
  const unsigned short* base = A + row * (size_t)ldk + (size_t)k0 + 8 * half;
  f.q[0] = *(const uint4*)(base);
  f.q[1] = *(const uint4*)(base + 16);
  return f.v;
}

// B fragment (32x16 bf16, KxN): lane L -> col N = L%16, half covers
// K in [8h, 8h+8) and [16+8h, 16+8h+8) (column-major B == row n of BT[N,K]).
__device__ __forceinline__ v16bf load_b_frag(const unsigned short* BT, size_t col,
                                             int ldk, int k0, int half) {
  FragBF f;
  const unsigned short* base = BT + col * (size_t)ldk + (size_t)k0 + 8 * half;
  f.q[0] = *(const uint4*)(base);
  f.q[1] = *(const uint4*)(base + 16);
  return f.v;
}

#define WMMA_BF16(A_, B_, C_) \
  __builtin_amdgcn_wmma_f32_16x16x32_bf16(false, (A_), false, (B_), (short)0, (C_), false, false)

// ---------------- generic GEMM: C[M,N] = A[M,K](bf16) @ BT[N,K]^T(bf16) ----------------
// block = 128 threads (4 waves); block tile 128x64. Each wave owns 2 M-tiles
// (rows wave*16 and 64+wave*16) x 4 N-tiles: 8 WMMAs per 12 b128 loads per
// k-step. 2-phase software pipeline with two named fragment sets so the
// rotation costs zero register moves. Requires K % 64 == 0 (holds here).
__global__ __launch_bounds__(128) void gemm_bf16_kernel(
    const unsigned short* __restrict__ A, const unsigned short* __restrict__ BT,
    const float* __restrict__ bias, float* __restrict__ Cf,
    unsigned short* __restrict__ Cbf, int N, int K, int relu)
{
  const int lane = threadIdx.x & 31;
  const int wave = threadIdx.x >> 5;
  const int half = lane >> 4;
  const int l16  = lane & 15;
  const size_t m0 = (size_t)blockIdx.y * 128 + (size_t)wave * 16;  // M-tile 0
  const size_t m1 = m0 + 64;                                       // M-tile 1
  const size_t nbase = (size_t)blockIdx.x * 64;

  v8f acc[2][4] = {};

  v16bf a0s0, a1s0, bs0[4];   // fragment set S0
  v16bf a0s1, a1s1, bs1[4];   // fragment set S1

  auto load_set = [&](v16bf& A0, v16bf& A1, v16bf* B, int kk) {
    A0 = load_a_frag(A, m0 + l16, K, kk, half);
    A1 = load_a_frag(A, m1 + l16, K, kk, half);
#pragma unroll
    for (int nt = 0; nt < 4; ++nt)
      B[nt] = load_b_frag(BT, nbase + nt * 16 + l16, K, kk, half);
  };
  auto mma_set = [&](const v16bf& A0, const v16bf& A1, const v16bf* B) {
#pragma unroll
    for (int nt = 0; nt < 4; ++nt) {
      acc[0][nt] = WMMA_BF16(A0, B[nt], acc[0][nt]);
      acc[1][nt] = WMMA_BF16(A1, B[nt], acc[1][nt]);
    }
  };

  load_set(a0s0, a1s0, bs0, 0);
  int k0 = 0;
  for (; k0 + 64 < K; k0 += 64) {
    load_set(a0s1, a1s1, bs1, k0 + 32);      // refill S1 while S0 computes
    if (k0 + 96 < K)
      __builtin_prefetch(BT + (nbase + l16) * (size_t)K + k0 + 96, 0, 3);
    mma_set(a0s0, a1s0, bs0);
    load_set(a0s0, a1s0, bs0, k0 + 64);      // refill S0 while S1 computes
    mma_set(a0s1, a1s1, bs1);
  }
  load_set(a0s1, a1s1, bs1, k0 + 32);        // tail pair
  mma_set(a0s0, a1s0, bs0);
  mma_set(a0s1, a1s1, bs1);

  // C/D layout: lane -> col n = l16, rows m = 8*half + r (r = VGPR index)
#pragma unroll
  for (int mt = 0; mt < 2; ++mt) {
    size_t mb = (mt == 0) ? m0 : m1;
#pragma unroll
    for (int nt = 0; nt < 4; ++nt) {
      size_t n = nbase + (size_t)nt * 16 + l16;
      float bv = bias ? bias[n] : 0.0f;
#pragma unroll
      for (int r = 0; r < 8; ++r) {
        size_t m = mb + (size_t)half * 8 + r;
        float v = acc[mt][nt][r] + bv;
        if (relu) v = fmaxf(v, 0.0f);
        if (Cf)  Cf[m * (size_t)N + n] = v;
        if (Cbf) Cbf[m * (size_t)N + n] = f2bf(v);
      }
    }
  }
}

// ---------------- GRU recurrent step ----------------
// grid = H/64 blocks, block = 128 threads (4 waves). Block owns 64 H-columns.
// gh[g] = h_prev[32,H] @ whh[g*H+j, :]^T  via WMMA (wave w = 16-col strip,
// A shared across the 3 gates), same 2-phase pipelined fragment loads, then
// f32 gate math with precomputed gi (= x@wih.T + bih). Requires H % 64 == 0.
__global__ __launch_bounds__(128) void gru_step_kernel(
    const float* __restrict__ Gi,            // [GG, 3H], row = b*TLEN + t
    const unsigned short* __restrict__ Whh,  // [3H, H] bf16
    const float* __restrict__ bhh,           // [3H]
    const float* __restrict__ hprev32, const unsigned short* __restrict__ hprevbf,
    float* __restrict__ hnext32, unsigned short* __restrict__ hnextbf,
    unsigned short* __restrict__ Yout,       // [GG, H] bf16 sequence out
    int t, int H)
{
  __shared__ float sgh[3][BS][64];
  const int lane = threadIdx.x & 31;
  const int wave = threadIdx.x >> 5;
  const int half = lane >> 4;
  const int l16  = lane & 15;
  const int jblk = blockIdx.x * 64;
  const size_t col = (size_t)jblk + wave * 16 + l16;

  v8f acc[3][2] = {};

  v16bf a0s0, a1s0, bs0[3];
  v16bf a0s1, a1s1, bs1[3];

  auto load_set = [&](v16bf& A0, v16bf& A1, v16bf* B, int kk) {
    A0 = load_a_frag(hprevbf, (size_t)l16,      H, kk, half);   // rows 0..15
    A1 = load_a_frag(hprevbf, (size_t)16 + l16, H, kk, half);   // rows 16..31
#pragma unroll
    for (int g = 0; g < 3; ++g)
      B[g] = load_b_frag(Whh, (size_t)g * H + col, H, kk, half);
  };
  auto mma_set = [&](const v16bf& A0, const v16bf& A1, const v16bf* B) {
#pragma unroll
    for (int g = 0; g < 3; ++g) {
      acc[g][0] = WMMA_BF16(A0, B[g], acc[g][0]);
      acc[g][1] = WMMA_BF16(A1, B[g], acc[g][1]);
    }
  };

  load_set(a0s0, a1s0, bs0, 0);
  int k0 = 0;
  for (; k0 + 64 < H; k0 += 64) {
    load_set(a0s1, a1s1, bs1, k0 + 32);
    mma_set(a0s0, a1s0, bs0);
    load_set(a0s0, a1s0, bs0, k0 + 64);
    mma_set(a0s1, a1s1, bs1);
  }
  load_set(a0s1, a1s1, bs1, k0 + 32);
  mma_set(a0s0, a1s0, bs0);
  mma_set(a0s1, a1s1, bs1);

#pragma unroll
  for (int g = 0; g < 3; ++g)
#pragma unroll
    for (int mt = 0; mt < 2; ++mt)
#pragma unroll
      for (int r = 0; r < 8; ++r)
        sgh[g][mt * 16 + half * 8 + r][wave * 16 + l16] = acc[g][mt][r];
  __syncthreads();

  for (int i = threadIdx.x; i < BS * 64; i += 128) {
    int b_ = i >> 6, jl = i & 63;
    int j = jblk + jl;
    size_t row = (size_t)b_ * TLEN + t;
    const float* gi = Gi + row * (size_t)(3 * H);
    float r  = 1.0f / (1.0f + __expf(-(gi[j]         + sgh[0][b_][jl] + bhh[j])));
    float z  = 1.0f / (1.0f + __expf(-(gi[H + j]     + sgh[1][b_][jl] + bhh[H + j])));
    float nn = tanhf(gi[2 * H + j] + r * (sgh[2][b_][jl] + bhh[2 * H + j]));
    float hp = hprev32[b_ * H + j];
    float hv = (1.0f - z) * nn + z * hp;
    hnext32[b_ * H + j] = hv;
    unsigned short hb = f2bf(hv);
    hnextbf[b_ * H + j] = hb;
    Yout[row * (size_t)H + j] = hb;
  }
}

// ---------------- GCN ----------------
__global__ void gcn_norm_kernel(const int* __restrict__ ei, float* __restrict__ enorm,
                                float* __restrict__ snorm)
{
  __shared__ float deg[NNODE];
  __shared__ float dinv[NNODE];
  int tid = threadIdx.x;
  if (tid < NNODE) deg[tid] = 1.0f;
  __syncthreads();
  for (int e = tid; e < NEDGE; e += blockDim.x)
    atomicAdd(&deg[ei[NEDGE + e]], 1.0f);
  __syncthreads();
  if (tid < NNODE) {
    dinv[tid] = rsqrtf(deg[tid]);
  }
  __syncthreads();
  if (tid < NNODE) snorm[tid] = dinv[tid] * dinv[tid];
  for (int e = tid; e < NEDGE; e += blockDim.x)
    enorm[e] = dinv[ei[e]] * dinv[ei[NEDGE + e]];
}

// one block (64 threads = 1 node each) per graph; LDS message passing.
__global__ __launch_bounds__(64) void gcn_kernel(
    const float* __restrict__ x, const int* __restrict__ ei,
    const float* __restrict__ w1, const float* __restrict__ b1,
    const float* __restrict__ w2, const float* __restrict__ b2,
    const float* __restrict__ cw, const float* __restrict__ cb,
    const float* __restrict__ enorm, const float* __restrict__ snorm,
    unsigned short* __restrict__ res)
{
  __shared__ float sA[NNODE][4];   // agg1 -> hw2/agg2
  __shared__ float sB[NNODE][4];   // hw1 -> h1
  const int g = blockIdx.x;
  const int n = threadIdx.x;

  float x0 = x[(size_t)g * NNODE * 2 + n * 2 + 0];
  float x1 = x[(size_t)g * NNODE * 2 + n * 2 + 1];
  // conv1: hw1 = x @ W1  (W1 [2,4] row-major)
#pragma unroll
  for (int f = 0; f < 4; ++f) sB[n][f] = x0 * w1[f] + x1 * w1[4 + f];
  float sn = snorm[n];
#pragma unroll
  for (int f = 0; f < 4; ++f) sA[n][f] = sB[n][f] * sn;   // self-loop term
  __syncthreads();
  for (int e = n; e < NEDGE; e += NNODE) {
    int s = ei[e], d = ei[NEDGE + e];
    float nm = enorm[e];
#pragma unroll
    for (int f = 0; f < 4; ++f) atomicAdd(&sA[d][f], sB[s][f] * nm);
  }
  __syncthreads();
  float h1[4];
#pragma unroll
  for (int f = 0; f < 4; ++f) h1[f] = tanhf(sA[n][f] + b1[f]);
  __syncthreads();
#pragma unroll
  for (int f = 0; f < 4; ++f) sB[n][f] = h1[f];
  // conv2: hw2 = h1 @ W2  (W2 [4,2])
  float hw2[2];
#pragma unroll
  for (int c = 0; c < 2; ++c)
    hw2[c] = h1[0] * w2[c] + h1[1] * w2[2 + c] + h1[2] * w2[4 + c] + h1[3] * w2[6 + c];
  __syncthreads();
#pragma unroll
  for (int c = 0; c < 2; ++c) { sA[n][c] = hw2[c]; sA[n][2 + c] = hw2[c] * sn; }
  __syncthreads();
  for (int e = n; e < NEDGE; e += NNODE) {
    int s = ei[e], d = ei[NEDGE + e];
    float nm = enorm[e];
#pragma unroll
    for (int c = 0; c < 2; ++c) atomicAdd(&sA[d][2 + c], sA[s][c] * nm);
  }
  __syncthreads();
  float h2[2];
#pragma unroll
  for (int c = 0; c < 2; ++c) h2[c] = tanhf(sA[n][2 + c] + b2[c]);
  // classifier: out = h2 @ cw + cb  (cw [2,2])
#pragma unroll
  for (int c = 0; c < 2; ++c) {
    float o = h2[0] * cw[c] + h2[1] * cw[2 + c] + cb[c];
    res[(size_t)g * INF_ + n * 2 + c] = f2bf(o);
  }
}

// ---------------- utilities ----------------
__global__ void f32_to_bf16_kernel(const float* __restrict__ in,
                                   unsigned short* __restrict__ out, size_t n) {
  size_t i = (size_t)blockIdx.x * blockDim.x + threadIdx.x;
  if (i < n) out[i] = f2bf(in[i]);
}

__global__ void transpose_f32_to_bf16_kernel(const float* __restrict__ in,
                                             unsigned short* __restrict__ out,
                                             int R, int C) {
  size_t i = (size_t)blockIdx.x * blockDim.x + threadIdx.x;
  size_t total = (size_t)R * C;
  if (i < total) {
    size_t r = i / C, c = i % C;
    out[c * (size_t)R + r] = f2bf(in[i]);
  }
}

__global__ void zero_h_kernel(float* a, float* b, unsigned short* c,
                              unsigned short* d, int n) {
  int i = blockIdx.x * blockDim.x + threadIdx.x;
  if (i < n) { a[i] = 0.0f; b[i] = 0.0f; c[i] = 0; d[i] = 0; }
}

// ---------------- host orchestration ----------------
extern "C" void kernel_launch(void* const* d_in, const int* in_sizes, int n_in,
                              void* d_out, int out_size, void* d_ws, size_t ws_size,
                              hipStream_t stream) {
  (void)in_sizes; (void)n_in; (void)out_size; (void)ws_size;
  const float* x        = (const float*)d_in[0];
  const int*   ei       = (const int*)d_in[1];
  const float* gcn_w1   = (const float*)d_in[2];
  const float* gcn_b1   = (const float*)d_in[3];
  const float* gcn_w2   = (const float*)d_in[4];
  const float* gcn_b2   = (const float*)d_in[5];
  const float* gcn_cw   = (const float*)d_in[6];
  const float* gcn_cb   = (const float*)d_in[7];
  const float* enc_fl_w = (const float*)d_in[8];
  const float* enc_fl_b = (const float*)d_in[9];
  const float* enc_wih  = (const float*)d_in[10];
  const float* enc_whh  = (const float*)d_in[11];
  const float* enc_bih  = (const float*)d_in[12];
  const float* enc_bhh  = (const float*)d_in[13];
  const float* dec_fl_w = (const float*)d_in[14];
  const float* dec_fl_b = (const float*)d_in[15];
  const float* dec_wih0 = (const float*)d_in[16];
  const float* dec_wih1 = (const float*)d_in[17];
  const float* dec_whh  = (const float*)d_in[18];
  const float* dec_bih  = (const float*)d_in[19];
  const float* dec_bhh  = (const float*)d_in[20];
  const float* dec_lin_w = (const float*)d_in[21];
  const float* dec_lin_b = (const float*)d_in[22];
  float* out = (float*)d_out;

  // -------- workspace carve-out --------
  char* wsp = (char*)d_ws;
  size_t off = 0;
  auto take = [&](size_t bytes) -> void* {
    void* r = wsp + off;
    off = (off + bytes + 255) & ~(size_t)255;
    return r;
  };
  unsigned short* enc_fl_wt  = (unsigned short*)take((size_t)H8 * INF_ * 2);        // [1024,128]
  unsigned short* enc_wih_bf = (unsigned short*)take((size_t)2 * 3 * H8 * H8 * 2);  // [2,3072,1024]
  unsigned short* enc_whh_bf = (unsigned short*)take((size_t)2 * 3 * H8 * H8 * 2);
  unsigned short* dec_fl_wt  = (unsigned short*)take((size_t)H8 * H8 * 2);          // [1024,1024]
  unsigned short* dec_wih0_bf = (unsigned short*)take((size_t)3 * HOUT * H8 * 2);   // [384,1024]
  unsigned short* dec_wih1_bf = (unsigned short*)take((size_t)3 * HOUT * HOUT * 2); // [384,128]
  unsigned short* dec_whh_bf = (unsigned short*)take((size_t)2 * 3 * HOUT * HOUT * 2);
  unsigned short* dec_lin_wt = (unsigned short*)take((size_t)HOUT * HOUT * 2);      // [128,128]
  unsigned short* res_bf = (unsigned short*)take((size_t)GG * INF_ * 2);
  unsigned short* seqA   = (unsigned short*)take((size_t)GG * H8 * 2);
  unsigned short* seqB   = (unsigned short*)take((size_t)GG * H8 * 2);
  float* Gi    = (float*)take((size_t)GG * 3 * H8 * 4);
  float* h32a  = (float*)take((size_t)BS * H8 * 4);
  float* h32b  = (float*)take((size_t)BS * H8 * 4);
  unsigned short* hbfa = (unsigned short*)take((size_t)BS * H8 * 2);
  unsigned short* hbfb = (unsigned short*)take((size_t)BS * H8 * 2);
  float* enorm = (float*)take(NEDGE * 4);
  float* snorm = (float*)take(NNODE * 4);

  auto cvt = [&](const float* src, unsigned short* dst, size_t n) {
    f32_to_bf16_kernel<<<(unsigned)((n + 255) / 256), 256, 0, stream>>>(src, dst, n);
  };
  auto tpose = [&](const float* src, unsigned short* dst, int R, int C) {
    size_t n = (size_t)R * C;
    transpose_f32_to_bf16_kernel<<<(unsigned)((n + 255) / 256), 256, 0, stream>>>(src, dst, R, C);
  };
  // GEMM: C[M,N] = A[M,K] @ BT[N,K]^T ; M fixed at GG, block tile 128x64
  auto gemm = [&](const unsigned short* A, const unsigned short* BT, const float* bias,
                  float* Cf, unsigned short* Cbf, int N, int K, int relu) {
    gemm_bf16_kernel<<<dim3((unsigned)(N / 64), (unsigned)(GG / 128)), 128, 0, stream>>>(
        A, BT, bias, Cf, Cbf, N, K, relu);
  };

  // -------- weight prep (bf16, B^T layouts) --------
  cvt(enc_wih, enc_wih_bf, (size_t)2 * 3 * H8 * H8);
  cvt(enc_whh, enc_whh_bf, (size_t)2 * 3 * H8 * H8);
  cvt(dec_wih0, dec_wih0_bf, (size_t)3 * HOUT * H8);
  cvt(dec_wih1, dec_wih1_bf, (size_t)3 * HOUT * HOUT);
  cvt(dec_whh, dec_whh_bf, (size_t)2 * 3 * HOUT * HOUT);
  tpose(enc_fl_w, enc_fl_wt, INF_, H8);
  tpose(dec_fl_w, dec_fl_wt, H8, H8);
  tpose(dec_lin_w, dec_lin_wt, HOUT, HOUT);

  // -------- GCN --------
  gcn_norm_kernel<<<1, 128, 0, stream>>>(ei, enorm, snorm);
  gcn_kernel<<<GG, NNODE, 0, stream>>>(x, ei, gcn_w1, gcn_b1, gcn_w2, gcn_b2,
                                       gcn_cw, gcn_cb, enorm, snorm, res_bf);

  // -------- encoder feature layer: relu(res @ enc_fl_w + b) -> seqA (bf16) --------
  gemm(res_bf, enc_fl_wt, enc_fl_b, nullptr, seqA, H8, INF_, 1);

  // -------- GRU driver --------
  auto run_gru = [&](const unsigned short* Xseq, int Kin,
                     const unsigned short* wih_bf, const float* bih,
                     const unsigned short* whh_bf, const float* bhh,
                     int H, unsigned short* Yseq) {
    // gi for all (b,t) in one GEMM: [GG,Kin] @ [Kin,3H] + bih
    gemm(Xseq, wih_bf, bih, Gi, nullptr, 3 * H, Kin, 0);
    int hn = BS * H;
    zero_h_kernel<<<(hn + 255) / 256, 256, 0, stream>>>(h32a, h32b, hbfa, hbfb, hn);
    int p = 0;
    for (int t = 0; t < TLEN; ++t) {
      gru_step_kernel<<<H / 64, 128, 0, stream>>>(
          Gi, whh_bf, bhh,
          p ? h32b : h32a, p ? hbfb : hbfa,
          p ? h32a : h32b, p ? hbfa : hbfb,
          Yseq, t, H);
      p ^= 1;
    }
  };

  // encoder GRU (2 layers, H=1024)
  run_gru(seqA, H8, enc_wih_bf, enc_bih,
          enc_whh_bf, enc_bhh, H8, seqB);
  run_gru(seqB, H8, enc_wih_bf + (size_t)3 * H8 * H8, enc_bih + 3 * H8,
          enc_whh_bf + (size_t)3 * H8 * H8, enc_bhh + 3 * H8, H8, seqA);

  // decoder feature layer: relu(seqA @ dec_fl_w + b) -> seqB
  gemm(seqA, dec_fl_wt, dec_fl_b, nullptr, seqB, H8, H8, 1);

  // decoder GRU layer 0 (Kin=1024 -> H=128), layer 1 (Kin=128 -> H=128)
  run_gru(seqB, H8, dec_wih0_bf, dec_bih,
          dec_whh_bf, dec_bhh, HOUT, seqA);
  run_gru(seqA, HOUT, dec_wih1_bf, dec_bih + 3 * HOUT,
          dec_whh_bf + (size_t)3 * HOUT * HOUT, dec_bhh + 3 * HOUT, HOUT, seqB);

  // final linear: seqB[GG,128] @ dec_lin_w + b -> d_out (f32)
  gemm(seqB, dec_lin_wt, dec_lin_b, out, nullptr, HOUT, HOUT, 0);
}
